// AFBlock_73942156968234
// MI455X (gfx1250) — compile-verified, matrix operand order checked by reading, not run on hardware
//
#include <hip/hip_runtime.h>
#include <cstdint>

// ---------------------------------------------------------------------------
// Transformer encoder block for MI455X (gfx1250, wave32, WMMA + TDM).
// GEMMs: V_WMMA_F32_16X16X32_BF16, LDS staged via TENSOR_LOAD_TO_LDS (TDM),
// double-buffered so DMA overlaps the matrix pipes.
// ---------------------------------------------------------------------------

typedef __bf16 bf16;
typedef __attribute__((ext_vector_type(16))) __bf16 v16bf;
typedef __attribute__((ext_vector_type(8)))  __bf16 bf16x8;
typedef __attribute__((ext_vector_type(8)))  float  v8f;
typedef __attribute__((ext_vector_type(4)))  unsigned int u32x4;
typedef __attribute__((ext_vector_type(8)))  int i32x8;
typedef __attribute__((ext_vector_type(4)))  int i32x4;

union U8F  { v8f v; float f[8]; };
union U16B { v16bf v; bf16x8 h[2]; bf16 e[16]; };

constexpr int cN = 8;
constexpr int cL = 1024;
constexpr int cE = 1024;
constexpr int cH = 8;
constexpr int cD = 128;   // head dim
constexpr int cF = 4096;

// ---------------------------------------------------------------------------
// TDM: DMA a 2-D tile (tileElems0 x tileRows, bf16) from row-major global
// memory into LDS, adding padAmount DWORDs of LDS padding after each
// (1<<padIntervalCode)*8 bytes (== one row here).  D# layout per ISA 8.3/8.4:
//   group0: [1:0]=count=1, [63:32]=lds_addr, [120:64]=global_addr, [127:126]=2
//   group1: data_size(2B)=1<<16, pad_enable<<20, pad_interval<<22,
//           pad_amount<<25, tensor_dim0/1 (huge: no OOB clamp),
//           tile_dim0/1, tensor_dim0_stride (elements). tile_dim2=0, s1=0.
//   groups 2/3 + extra group = 0 (2-D tile).
// Wave-level op (EXEC-independent), tracked with TENSORcnt.
// This toolchain exposes the 6-arg builtin (clang-23 / therock-10.0 form).
// ---------------------------------------------------------------------------
__device__ inline void tdm_load_2d(void* ldsDst, const void* gSrc,
                                   unsigned tileElems0, unsigned tileRows,
                                   unsigned long long strideElems0,
                                   unsigned padIntervalCode,
                                   unsigned padAmountCode) {
  const unsigned int lds = (unsigned int)(uintptr_t)ldsDst;
  const unsigned long long ga = (unsigned long long)(uintptr_t)gSrc;
  u32x4 g0;
  g0[0] = 1u;                                            // count=1 (valid)
  g0[1] = lds;
  g0[2] = (unsigned int)(ga & 0xFFFFFFFFull);
  g0[3] = (unsigned int)((ga >> 32) & 0x01FFFFFFull) | (2u << 30);  // type=2
  const unsigned dim0 = 0x40000000u, dim1 = 0x40000000u; // huge: no OOB clamp
  i32x8 g1;
  g1[0] = (int)((1u << 16) | (1u << 20) |                // 2B elems, pad on
                (padIntervalCode << 22) | (padAmountCode << 25));
  g1[1] = (int)((dim0 & 0xFFFFu) << 16);                 // tensor_dim0 lo
  g1[2] = (int)((dim0 >> 16) | ((dim1 & 0xFFFFu) << 16));
  g1[3] = (int)((dim1 >> 16) | (tileElems0 << 16));      // tile_dim0
  g1[4] = (int)(tileRows & 0xFFFFu);                     // tile_dim1 (dim2=0)
  g1[5] = (int)(unsigned)(strideElems0 & 0xFFFFFFFFull); // dim0 stride lo
  g1[6] = (int)(unsigned)((strideElems0 >> 32) & 0xFFFFull);
  g1[7] = 0;
  const i32x4 z4 = {0, 0, 0, 0};
  const i32x8 z8 = {0, 0, 0, 0, 0, 0, 0, 0};
  __builtin_amdgcn_tensor_load_to_lds(g0, g1, z4, z4, z8, 0);
}

// ---------------------------------------------------------------------------
// Fragment loader for 16-bit WMMA operands from a row-major LDS tile.
//   lanes 0-15 : row = rowBase+lane,  elems 0-7 = K 0..7,  8-15 = K 16..23
//   lanes16-31 : row = rowBase+l-16,  elems 0-7 = K 8..15, 8-15 = K 24..31
// ---------------------------------------------------------------------------
__device__ inline v16bf frag_ld(const bf16* base, int rowBase, int colBase,
                                int ldElems, int lane) {
  const int hl = lane >> 4;
  const bf16* p = base + (rowBase + (lane & 15)) * ldElems + colBase + hl * 8;
  U16B u;
  u.h[0] = *(const bf16x8*)(p);
  u.h[1] = *(const bf16x8*)(p + 16);
  return u.v;
}

__device__ inline float rmax16(float v) {
#pragma unroll
  for (int off = 1; off < 16; off <<= 1) v = fmaxf(v, __shfl_xor(v, off, 32));
  return v;
}
__device__ inline float rsum16(float v) {
#pragma unroll
  for (int off = 1; off < 16; off <<= 1) v += __shfl_xor(v, off, 32);
  return v;
}

// ---------------------------------------------------------------------------
// Elementwise helpers.
// ---------------------------------------------------------------------------
__global__ void f32_to_bf16_kernel(const float* __restrict__ in,
                                   bf16* __restrict__ out, size_t n) {
  size_t i = (size_t)blockIdx.x * blockDim.x + threadIdx.x;
  size_t stride = (size_t)gridDim.x * blockDim.x;
  for (; i < n; i += stride) out[i] = (bf16)in[i];
}

__global__ void zero_f32_kernel(float* __restrict__ out, size_t n) {
  size_t i = (size_t)blockIdx.x * blockDim.x + threadIdx.x;
  size_t stride = (size_t)gridDim.x * blockDim.x;
  for (; i < n; i += stride) out[i] = 0.0f;
}

// ---------------------------------------------------------------------------
// GEMM: C[M,N] = A[M,K] @ W[N,K]^T + bias, optional ReLU. bf16 in, f32 accum.
// Block: 256 threads (8 waves). Block tile 128x64, wave tile 32x32.
// Staging: TDM descriptors issued by wave 0, double-buffered; TDM ops from
// one wave complete in order, so tensorcnt<=2 after issuing the next pair
// guarantees the current pair has landed while the next stays in flight.
// ---------------------------------------------------------------------------
__global__ __launch_bounds__(256) void gemm_bf16_nt_kernel(
    const bf16* __restrict__ A, const bf16* __restrict__ W,
    const float* __restrict__ bias, float* __restrict__ Cf,
    bf16* __restrict__ Cb, int M, int N, int K, int relu) {
  constexpr int BM = 128, BN = 64, BK = 32, LD = BK + 8;  // ld=40 -> 80B rows
  __shared__ bf16 As[2][BM * LD];
  __shared__ bf16 Bs[2][BN * LD];

  const int tid  = threadIdx.x;
  const int lane = tid & 31;
  const int wave = tid >> 5;     // 0..7
  const int wm   = wave >> 1;    // 0..3 : 32-row group
  const int wn   = wave & 1;     // 0..1 : 32-col group
  const int bm0  = blockIdx.y * BM;
  const int bn0  = blockIdx.x * BN;

  U8F acc[2][2];
#pragma unroll
  for (int i = 0; i < 2; i++)
#pragma unroll
    for (int j = 0; j < 2; j++)
#pragma unroll
      for (int r = 0; r < 8; r++) acc[i][j].f[r] = 0.0f;

  const int T = K / BK;
  // Prologue: DMA tile 0 into buffer 0 (row = 64B -> interval code 3;
  // pad 16B = 4 DWORDs -> amount code 3; gives LD=40).
  if (wave == 0) {
    tdm_load_2d(&As[0][0], &A[(size_t)bm0 * K], BK, BM, (unsigned)K, 3, 3);
    tdm_load_2d(&Bs[0][0], &W[(size_t)bn0 * K], BK, BN, (unsigned)K, 3, 3);
  }

  for (int t = 0; t < T; t++) {
    const int cur = t & 1;
    if (wave == 0) {
      if (t + 1 < T) {
        const int k0 = (t + 1) * BK;
        tdm_load_2d(&As[cur ^ 1][0], &A[(size_t)bm0 * K + k0], BK, BM,
                    (unsigned)K, 3, 3);
        tdm_load_2d(&Bs[cur ^ 1][0], &W[(size_t)bn0 * K + k0], BK, BN,
                    (unsigned)K, 3, 3);
        __builtin_amdgcn_s_wait_tensorcnt((short)2);  // pair t done
      } else {
        __builtin_amdgcn_s_wait_tensorcnt((short)0);
      }
    }
    __syncthreads();

    v16bf a0 = frag_ld(As[cur], wm * 32 + 0,  0, LD, lane);
    v16bf a1 = frag_ld(As[cur], wm * 32 + 16, 0, LD, lane);
    v16bf b0 = frag_ld(Bs[cur], wn * 32 + 0,  0, LD, lane);
    v16bf b1 = frag_ld(Bs[cur], wn * 32 + 16, 0, LD, lane);
    acc[0][0].v = __builtin_amdgcn_wmma_f32_16x16x32_bf16(
        false, a0, false, b0, (short)0, acc[0][0].v, false, false);
    acc[0][1].v = __builtin_amdgcn_wmma_f32_16x16x32_bf16(
        false, a0, false, b1, (short)0, acc[0][1].v, false, false);
    acc[1][0].v = __builtin_amdgcn_wmma_f32_16x16x32_bf16(
        false, a1, false, b0, (short)0, acc[1][0].v, false, false);
    acc[1][1].v = __builtin_amdgcn_wmma_f32_16x16x32_bf16(
        false, a1, false, b1, (short)0, acc[1][1].v, false, false);
    __syncthreads();  // all waves done reading buf[cur] before it is re-DMAed
  }

  // Epilogue. C/D layout: lane = column, vgpr r = row (r+8 for lanes 16-31).
  const int hl = lane >> 4;
  const int lc = lane & 15;
#pragma unroll
  for (int mt = 0; mt < 2; mt++) {
#pragma unroll
    for (int nt = 0; nt < 2; nt++) {
#pragma unroll
      for (int r = 0; r < 8; r++) {
        int row = bm0 + wm * 32 + mt * 16 + hl * 8 + r;
        int col = bn0 + wn * 32 + nt * 16 + lc;
        float v = acc[mt][nt].f[r];
        if (bias) v += bias[col];
        if (relu) v = fmaxf(v, 0.0f);
        if (Cf) Cf[(size_t)row * N + col] = v;
        if (Cb) Cb[(size_t)row * N + col] = (bf16)v;
      }
    }
  }
}

// ---------------------------------------------------------------------------
// Attention pass 1: per-row running max M and sum S of exp(masked score - M).
// grid (L/128, H, N); block 256 (8 waves x 16 rows).
// K chunks (32 rows x 128 d, row 256B -> interval code 5) staged via TDM.
// ---------------------------------------------------------------------------
__global__ __launch_bounds__(256) void attn_pass1_kernel(
    const bf16* __restrict__ qkv,        // [N*L, 3E]
    const int*  __restrict__ amask,      // [N, L]
    float* __restrict__ Mrow, float* __restrict__ Srow) {  // [N*H*L]
  constexpr int KLD = cD + 8;            // 136 -> 272B rows (16B aligned)
  __shared__ bf16 Ks[32 * KLD];
  __shared__ int  Msk[32];

  const int lane = threadIdx.x & 31;
  const int wave = threadIdx.x >> 5;
  const int n = blockIdx.z, h = blockIdx.y;
  const int l0 = blockIdx.x * 128 + wave * 16;
  const int hl = lane >> 4, lc = lane & 15;

  // Preload q fragments for this wave's 16 rows, all 4 K-steps of D=128.
  v16bf qf[4];
  {
    const bf16* qbase =
        qkv + (size_t)(n * cL + l0 + lc) * (3 * cE) + h * cD + hl * 8;
#pragma unroll
    for (int kk = 0; kk < 4; kk++) {
      U16B u;
      u.h[0] = *(const bf16x8*)(qbase + kk * 32);
      u.h[1] = *(const bf16x8*)(qbase + kk * 32 + 16);
      qf[kk] = u.v;
    }
  }

  float runMax[8], runSum[8];
#pragma unroll
  for (int r = 0; r < 8; r++) { runMax[r] = -1e30f; runSum[r] = 0.0f; }
  const float scale = 0.08838834764831845f;  // 1/sqrt(128)

  for (int m0 = 0; m0 < cL; m0 += 32) {
    if (wave == 0) {
      tdm_load_2d(Ks, &qkv[(size_t)(n * cL + m0) * (3 * cE) + cE + h * cD],
                  cD, 32, 3 * cE, 5, 3);
      __builtin_amdgcn_s_wait_tensorcnt((short)0);
    }
    if (threadIdx.x < 32) Msk[threadIdx.x] = amask[n * cL + m0 + threadIdx.x];
    __syncthreads();

    U8F s0, s1;
#pragma unroll
    for (int r = 0; r < 8; r++) { s0.f[r] = 0.0f; s1.f[r] = 0.0f; }
#pragma unroll
    for (int kk = 0; kk < 4; kk++) {
      v16bf b0 = frag_ld(Ks, 0,  kk * 32, KLD, lane);
      v16bf b1 = frag_ld(Ks, 16, kk * 32, KLD, lane);
      s0.v = __builtin_amdgcn_wmma_f32_16x16x32_bf16(
          false, qf[kk], false, b0, (short)0, s0.v, false, false);
      s1.v = __builtin_amdgcn_wmma_f32_16x16x32_bf16(
          false, qf[kk], false, b1, (short)0, s1.v, false, false);
    }

    float sc[2][8];
#pragma unroll
    for (int ms = 0; ms < 2; ms++) {
      const U8F& a = ms ? s1 : s0;
      int col = m0 + ms * 16 + lc;
      bool km = (Msk[ms * 16 + lc] != 0);
#pragma unroll
      for (int r = 0; r < 8; r++) {
        int row = l0 + hl * 8 + r;
        bool keep = (row >= col) && km;
        sc[ms][r] = a.f[r] * scale + (keep ? 0.0f : -999.0f);
      }
    }
#pragma unroll
    for (int r = 0; r < 8; r++) {
      float tmax = rmax16(fmaxf(sc[0][r], sc[1][r]));
      float nm = fmaxf(runMax[r], tmax);
      float fac = __expf(runMax[r] - nm);
      float ts = rsum16(__expf(sc[0][r] - nm) + __expf(sc[1][r] - nm));
      runSum[r] = runSum[r] * fac + ts;
      runMax[r] = nm;
    }
    __syncthreads();
  }

  if (lc == 0) {
#pragma unroll
    for (int r = 0; r < 8; r++) {
      int row = l0 + hl * 8 + r;
      size_t idx = (size_t)(n * cH + h) * cL + row;
      Mrow[idx] = runMax[r];
      Srow[idx] = runSum[r];
    }
  }
}

// ---------------------------------------------------------------------------
// Attention pass 2: w = exp(score - M)/S, a += w/H (atomic), ctx = w @ v via
// WMMA (w re-fragmented through LDS). grid (L/128, H, N); block 256.
// ---------------------------------------------------------------------------
__global__ __launch_bounds__(256) void attn_pass2_kernel(
    const bf16* __restrict__ qkv, const int* __restrict__ amask,
    const float* __restrict__ Mrow, const float* __restrict__ Srow,
    bf16* __restrict__ ctx,        // [N*L, E]
    float* __restrict__ aout) {    // [N, L, L]
  constexpr int KLD = cD + 8;      // 136
  constexpr int WLD = 40;          // w tile ld (16x32 + pad)
  __shared__ bf16  Ks[32 * KLD];
  __shared__ bf16  Vs[32 * KLD];
  __shared__ bf16  Ws[8 * 16 * WLD];
  __shared__ int   Msk[32];
  __shared__ float Mb[128];
  __shared__ float Sb[128];

  const int lane = threadIdx.x & 31;
  const int wave = threadIdx.x >> 5;
  const int n = blockIdx.z, h = blockIdx.y;
  const int l0 = blockIdx.x * 128 + wave * 16;
  const int hl = lane >> 4, lc = lane & 15;

  if (threadIdx.x < 128) {
    size_t base = (size_t)(n * cH + h) * cL + blockIdx.x * 128 + threadIdx.x;
    Mb[threadIdx.x] = Mrow[base];
    Sb[threadIdx.x] = Srow[base];
  }

  v16bf qf[4];
  {
    const bf16* qbase =
        qkv + (size_t)(n * cL + l0 + lc) * (3 * cE) + h * cD + hl * 8;
#pragma unroll
    for (int kk = 0; kk < 4; kk++) {
      U16B u;
      u.h[0] = *(const bf16x8*)(qbase + kk * 32);
      u.h[1] = *(const bf16x8*)(qbase + kk * 32 + 16);
      qf[kk] = u.v;
    }
  }

  U8F cacc[8];  // ctx accumulators: 16 rows x 128 d
#pragma unroll
  for (int d = 0; d < 8; d++)
#pragma unroll
    for (int r = 0; r < 8; r++) cacc[d].f[r] = 0.0f;

  const float scale = 0.08838834764831845f;
  bf16* Wsw = Ws + wave * 16 * WLD;
  __syncthreads();  // Mb/Sb visible

  for (int m0 = 0; m0 < cL; m0 += 32) {
    if (wave == 0) {
      const bf16* base = &qkv[(size_t)(n * cL + m0) * (3 * cE) + h * cD];
      tdm_load_2d(Ks, base + cE,     cD, 32, 3 * cE, 5, 3);
      tdm_load_2d(Vs, base + 2 * cE, cD, 32, 3 * cE, 5, 3);
      __builtin_amdgcn_s_wait_tensorcnt((short)0);
    }
    if (threadIdx.x < 32) Msk[threadIdx.x] = amask[n * cL + m0 + threadIdx.x];
    __syncthreads();

    // Scores (16 l x 32 m) via 8 WMMAs over D.
    U8F s0, s1;
#pragma unroll
    for (int r = 0; r < 8; r++) { s0.f[r] = 0.0f; s1.f[r] = 0.0f; }
#pragma unroll
    for (int kk = 0; kk < 4; kk++) {
      v16bf b0 = frag_ld(Ks, 0,  kk * 32, KLD, lane);
      v16bf b1 = frag_ld(Ks, 16, kk * 32, KLD, lane);
      s0.v = __builtin_amdgcn_wmma_f32_16x16x32_bf16(
          false, qf[kk], false, b0, (short)0, s0.v, false, false);
      s1.v = __builtin_amdgcn_wmma_f32_16x16x32_bf16(
          false, qf[kk], false, b1, (short)0, s1.v, false, false);
    }

    // Normalize, emit a += w/H, stash w tile in LDS (per-wave region).
#pragma unroll
    for (int ms = 0; ms < 2; ms++) {
      const U8F& a = ms ? s1 : s0;
      int col = m0 + ms * 16 + lc;
      bool km = (Msk[ms * 16 + lc] != 0);
#pragma unroll
      for (int r = 0; r < 8; r++) {
        int rloc = hl * 8 + r;               // 0..15 within wave tile
        int row = l0 + rloc;                 // global l
        bool keep = (row >= col) && km;
        float s = a.f[r] * scale + (keep ? 0.0f : -999.0f);
        float w = __expf(s - Mb[wave * 16 + rloc]) / Sb[wave * 16 + rloc];
        atomicAdd(&aout[((size_t)n * cL + row) * cL + col], w * 0.125f);
        Wsw[rloc * WLD + ms * 16 + lc] = (bf16)w;
      }
    }

    // ctx += w(16x32) @ v(32x128): one WMMA per 16-wide d sub-tile.
    v16bf wa = frag_ld(Wsw, 0, 0, WLD, lane);
#pragma unroll
    for (int ds = 0; ds < 8; ds++) {
      U16B bu;
      int d = ds * 16 + lc;
#pragma unroll
      for (int e = 0; e < 8; e++) {
        bu.e[e]     = Vs[(hl * 8 + e) * KLD + d];        // K = hl*8 + e
        bu.e[e + 8] = Vs[(16 + hl * 8 + e) * KLD + d];   // K = 16 + hl*8 + e
      }
      cacc[ds].v = __builtin_amdgcn_wmma_f32_16x16x32_bf16(
          false, wa, false, bu.v, (short)0, cacc[ds].v, false, false);
    }
    __syncthreads();
  }

  // Store ctx (bf16) for the out-projection GEMM.
#pragma unroll
  for (int ds = 0; ds < 8; ds++) {
#pragma unroll
    for (int r = 0; r < 8; r++) {
      int row = l0 + hl * 8 + r;
      int d = ds * 16 + lc;
      ctx[(size_t)(n * cL + row) * cE + h * cD + d] = (bf16)cacc[ds].f[r];
    }
  }
}

// ---------------------------------------------------------------------------
// Fused residual add + LayerNorm over E=1024. Block per row, 256 threads.
// ---------------------------------------------------------------------------
__global__ __launch_bounds__(256) void add_ln_kernel(
    const float* __restrict__ X, const float* __restrict__ Y,
    const float* __restrict__ g, const float* __restrict__ b,
    float* __restrict__ outF, bf16* __restrict__ outB) {
  __shared__ float red[16];
  const size_t row = blockIdx.x;
  const float* x = X + row * cE;
  const float* y = Y + row * cE;
  float v[4], s = 0.0f, s2 = 0.0f;
#pragma unroll
  for (int i = 0; i < 4; i++) {
    int c = threadIdx.x + i * 256;
    float t = x[c] + y[c];
    v[i] = t; s += t; s2 += t * t;
  }
#pragma unroll
  for (int off = 16; off; off >>= 1) {
    s  += __shfl_xor(s, off, 32);
    s2 += __shfl_xor(s2, off, 32);
  }
  const int lane = threadIdx.x & 31, wave = threadIdx.x >> 5;
  if (lane == 0) { red[wave] = s; red[8 + wave] = s2; }
  __syncthreads();
  if (wave == 0) {
    s  = (lane < 8) ? red[lane] : 0.0f;
    s2 = (lane < 8) ? red[8 + lane] : 0.0f;
#pragma unroll
    for (int off = 4; off; off >>= 1) {
      s  += __shfl_xor(s, off, 32);
      s2 += __shfl_xor(s2, off, 32);
    }
    if (lane == 0) { red[0] = s; red[1] = s2; }
  }
  __syncthreads();
  const float mu = red[0] * (1.0f / cE);
  const float var = red[1] * (1.0f / cE) - mu * mu;
  const float rstd = rsqrtf(var + 1e-5f);
#pragma unroll
  for (int i = 0; i < 4; i++) {
    int c = threadIdx.x + i * 256;
    float t = (v[i] - mu) * rstd * g[c] + b[c];
    if (outF) outF[row * cE + c] = t;
    if (outB) outB[row * cE + c] = (bf16)t;
  }
}

// ---------------------------------------------------------------------------
// Host-side orchestration.
// ---------------------------------------------------------------------------
extern "C" void kernel_launch(void* const* d_in, const int* in_sizes, int n_in,
                              void* d_out, int out_size, void* d_ws,
                              size_t ws_size, hipStream_t stream) {
  const float* x    = (const float*)d_in[0];
  const int*   am   = (const int*)d_in[1];
  const float* wqkv = (const float*)d_in[2];
  const float* bqkv = (const float*)d_in[3];
  const float* wout = (const float*)d_in[4];
  const float* bout = (const float*)d_in[5];
  const float* ln1g = (const float*)d_in[6];
  const float* ln1b = (const float*)d_in[7];
  const float* ln2g = (const float*)d_in[8];
  const float* ln2b = (const float*)d_in[9];
  const float* w1   = (const float*)d_in[10];
  const float* b1   = (const float*)d_in[11];
  const float* w2   = (const float*)d_in[12];
  const float* b2   = (const float*)d_in[13];

  const size_t NL = (size_t)cN * cL;   // 8192 rows
  float* z_out = (float*)d_out;                 // (N,L,E) = 8M floats
  float* a_out = (float*)d_out + NL * cE;       // (N,L,L) = 8M floats

  // Workspace carve-up (~300 MB total).
  char* ws = (char*)d_ws;
  size_t off = 0;
  auto alloc = [&](size_t bytes) -> void* {
    void* p = ws + off;
    off += (bytes + 255) & ~((size_t)255);
    return p;
  };
  bf16*  x_bf    = (bf16*)alloc(NL * cE * 2);
  bf16*  wqkv_bf = (bf16*)alloc((size_t)3 * cE * cE * 2);
  bf16*  wout_bf = (bf16*)alloc((size_t)cE * cE * 2);
  bf16*  w1_bf   = (bf16*)alloc((size_t)cF * cE * 2);
  bf16*  w2_bf   = (bf16*)alloc((size_t)cE * cF * 2);
  bf16*  qkv_bf  = (bf16*)alloc(NL * 3 * cE * 2);
  bf16*  ctx_bf  = (bf16*)alloc(NL * cE * 2);
  float* attnout = (float*)alloc(NL * cE * 4);
  float* z1_f    = (float*)alloc(NL * cE * 4);
  bf16*  z1_bf   = (bf16*)alloc(NL * cE * 2);
  bf16*  ff_bf   = (bf16*)alloc(NL * cF * 2);
  float* ff2_f   = (float*)alloc(NL * cE * 4);
  float* Mrow    = (float*)alloc((size_t)cN * cH * cL * 4);
  float* Srow    = (float*)alloc((size_t)cN * cH * cL * 4);
  (void)ws_size; (void)n_in; (void)in_sizes; (void)out_size;

  // 1) bf16 conversions of x and weights.
  f32_to_bf16_kernel<<<2048, 256, 0, stream>>>(x, x_bf, NL * cE);
  f32_to_bf16_kernel<<<2048, 256, 0, stream>>>(wqkv, wqkv_bf, (size_t)3 * cE * cE);
  f32_to_bf16_kernel<<<1024, 256, 0, stream>>>(wout, wout_bf, (size_t)cE * cE);
  f32_to_bf16_kernel<<<2048, 256, 0, stream>>>(w1, w1_bf, (size_t)cF * cE);
  f32_to_bf16_kernel<<<2048, 256, 0, stream>>>(w2, w2_bf, (size_t)cE * cF);
  // 2) zero the attention-average output (accumulated via atomics).
  zero_f32_kernel<<<2048, 256, 0, stream>>>(a_out, NL * cL);

  // 3) QKV projection: qkv = x @ in_proj_w^T + in_proj_b  (bf16 out).
  gemm_bf16_nt_kernel<<<dim3(3 * cE / 64, NL / 128), 256, 0, stream>>>(
      x_bf, wqkv_bf, bqkv, nullptr, qkv_bf, (int)NL, 3 * cE, cE, 0);

  // 4-5) Attention: row stats, then normalize + a-output + ctx.
  attn_pass1_kernel<<<dim3(cL / 128, cH, cN), 256, 0, stream>>>(
      qkv_bf, am, Mrow, Srow);
  attn_pass2_kernel<<<dim3(cL / 128, cH, cN), 256, 0, stream>>>(
      qkv_bf, am, Mrow, Srow, ctx_bf, a_out);

  // 6) Out projection: attnout = ctx @ out_w^T + out_b  (fp32 out).
  gemm_bf16_nt_kernel<<<dim3(cE / 64, NL / 128), 256, 0, stream>>>(
      ctx_bf, wout_bf, bout, attnout, nullptr, (int)NL, cE, cE, 0);

  // 7) z1 = LN(x + attnout).
  add_ln_kernel<<<NL, 256, 0, stream>>>(x, attnout, ln1g, ln1b, z1_f, z1_bf);

  // 8) ff = relu(z1 @ w1^T + b1)  (bf16 out).
  gemm_bf16_nt_kernel<<<dim3(cF / 64, NL / 128), 256, 0, stream>>>(
      z1_bf, w1_bf, b1, nullptr, ff_bf, (int)NL, cF, cE, 1);

  // 9) ff2 = ff @ w2^T + b2  (fp32 out).
  gemm_bf16_nt_kernel<<<dim3(cE / 64, NL / 128), 256, 0, stream>>>(
      ff_bf, w2_bf, b2, ff2_f, nullptr, (int)NL, cE, cF, 0);

  // 10) z = LN(z1 + ff2) -> first half of d_out.
  add_ln_kernel<<<NL, 256, 0, stream>>>(z1_f, ff2_f, ln2g, ln2b, z_out, nullptr);
}
